// VoiceCloningGenerator_47115791237345
// MI455X (gfx1250) — matrix operand order repaired and checked by reading, not compiled, not run on hardware
//
#include <hip/hip_runtime.h>
#include <hip/hip_bf16.h>

// ---------------------------------------------------------------------------
// Voice-cloning generator (Tacotron-like) for MI455X / gfx1250.
// - All GEMMs: v_wmma_f32_16x16x32_f16 (f16 in, f32 accumulate)
// - Weights pre-packed into ISA B-fragment lane layout (1 coalesced 32B/lane
//   load per 32x16 K-tile)
// - Decoder query tile staged LDS-side by the Tensor Data Mover
//   (tensor_load_to_lds + s_wait_tensorcnt), prefetched one step ahead.
// ---------------------------------------------------------------------------

typedef _Float16 v16h __attribute__((ext_vector_type(16)));
typedef _Float16 h8   __attribute__((ext_vector_type(8)));
typedef float    v8f  __attribute__((ext_vector_type(8)));
typedef unsigned int v4u __attribute__((ext_vector_type(4)));
typedef int      v8i  __attribute__((ext_vector_type(8)));
typedef int      v4i  __attribute__((ext_vector_type(4)));

#define B_  32
#define T_  256
#define H_  512
#define HE_ 256
#define E_  256
#define M_  80
#define S_  128

__device__ __forceinline__ v8f wmma16(v16h a, v16h b, v8f c) {
  // D = A(16x32 f16) * B(32x16 f16) + C(16x16 f32)
  return __builtin_amdgcn_wmma_f32_16x16x32_f16(false, a, false, b, (short)0, c,
                                                false, false);
}

// B fragment: pre-swizzled tiles, 512 halves (1KB) per 32x16 K-tile.
__device__ __forceinline__ v16h load_bfrag(const _Float16* __restrict__ W,
                                           int Ntiles, int kt, int nt, int lane) {
  const v16h* p = (const v16h*)(W + (((size_t)kt * Ntiles + nt) << 9));
  return p[lane];
}

// A fragment (16x32) from an f32 row-major matrix (LDS or global):
// lane ln -> row (ln&15); halves 0..7 = K kb..kb+7, halves 8..15 = kb+16..kb+23,
// kb = kt*32 + ((ln>>4)<<3).  (ISA 16-bit A layout.)
__device__ __forceinline__ v16h afrag_f32(const float* __restrict__ rowbase, int kb) {
  v16h a;
#pragma unroll
  for (int i = 0; i < 8; ++i) a[i]     = (_Float16)rowbase[kb + i];
#pragma unroll
  for (int i = 0; i < 8; ++i) a[8 + i] = (_Float16)rowbase[kb + 16 + i];
  return a;
}

__device__ __forceinline__ v16h afrag_f16(const _Float16* __restrict__ rowbase, int kb) {
  h8 lo = *(const h8*)(rowbase + kb);
  h8 hi = *(const h8*)(rowbase + kb + 16);
  v16h a;
#pragma unroll
  for (int i = 0; i < 8; ++i) { a[i] = lo[i]; a[8 + i] = hi[i]; }
  return a;
}

__device__ __forceinline__ float sigm(float x) { return 1.f / (1.f + __expf(-x)); }
__device__ __forceinline__ float tanh_fast(float x) {
  x = fminf(15.f, fmaxf(-15.f, x));
  float e = __expf(2.f * x);
  return (e - 1.f) / (e + 1.f);
}

// ---------------------------------------------------------------------------
// TDM: 2-D tile load Global -> LDS.  rows x rowLenElts of 4-byte elements,
// row stride rowStrideElts; written contiguously at lds_addr.
// D# bitfields per CDNA5 ISA 8.3/8.4 (group0: count/lds/global/type,
// group1: data_size + tensor/tile dims + dim0 stride). Groups 2/3 zero (2-D).
// This toolchain exposes the 6-arg builtin: (v4u, v8i, v4i, v4i, v8i, i32).
// ---------------------------------------------------------------------------
__device__ __forceinline__ void tdm_load_2d(unsigned lds_addr, const void* gaddr,
                                            unsigned rowLenElts, unsigned rows,
                                            unsigned rowStrideElts) {
  unsigned long long ga = (unsigned long long)(size_t)gaddr;
  v4u g0;
  g0[0] = 1u;                                          // count=1 (user D#)
  g0[1] = lds_addr;                                    // LDS byte address
  g0[2] = (unsigned)(ga & 0xFFFFFFFFu);                // global_addr[31:0]
  g0[3] = (unsigned)((ga >> 32) & 0x1FFFFFFu) | (2u << 30);  // [56:32] | type=2
  v8i g1;
  g1[0] = (int)(2u << 16);                             // data_size=2 (4B)
  g1[1] = (int)(rowLenElts << 16);                     // tensor_dim0[15:0]
  g1[2] = (int)((rowLenElts >> 16) & 0xFFFFu) | (int)(rows << 16);  // dim0 hi | dim1 lo
  g1[3] = (int)((rows >> 16) & 0xFFFFu) | (int)(rowLenElts << 16);  // dim1 hi | tile_dim0
  g1[4] = (int)rows;                                   // tile_dim1 (tile_dim2=0)
  g1[5] = (int)rowStrideElts;                          // tensor_dim0_stride[31:0]
  g1[6] = 0;                                           // stride[47:32] | dim1_stride lo
  g1[7] = 0;
  v4i z4 = {0, 0, 0, 0};
  v8i z8 = {0, 0, 0, 0, 0, 0, 0, 0};
  __builtin_amdgcn_tensor_load_to_lds(g0, g1, z4, z4, z8, 0);
}

// ---------------------------------------------------------------------------
// Weight pre-pack: src is torch-layout [Nsrc, srcStride] f32; logical K slice
// [kSrcOff, kSrcOff+Klen) lands at dst k-tile ktOff.  Dst region memset(0)
// first so K/N padding stays zero.
//   half r=ln*16+i of tile (kt,nt):  k = kt*32 + (ln<16?0:8) + (i<8?i:i+8),
//                                    n = nt*16 + (ln&15)
// ---------------------------------------------------------------------------
__global__ void pack_b_kernel(const float* __restrict__ src, int Nsrc, int srcStride,
                              int kSrcOff, int Klen,
                              _Float16* __restrict__ dst, int NtilesDst, int ktOff) {
  int NtCover = (Nsrc + 15) >> 4;
  int KtCover = (Klen + 31) >> 5;
  long total = (long)NtCover * KtCover * 512;
  for (long idx = (long)blockIdx.x * blockDim.x + threadIdx.x; idx < total;
       idx += (long)gridDim.x * blockDim.x) {
    int r = (int)(idx & 511);
    long tile = idx >> 9;
    int nt  = (int)(tile % NtCover);
    int ktL = (int)(tile / NtCover);
    int ln = r >> 4, i = r & 15;
    int k = ktL * 32 + ((ln & 16) ? 8 : 0) + (i < 8 ? i : i + 8);
    int n = nt * 16 + (ln & 15);
    if (k < Klen && n < Nsrc)
      dst[(((size_t)(ktOff + ktL)) * NtilesDst + nt) * 512 + r] =
          (_Float16)src[(size_t)n * srcStride + kSrcOff + k];
  }
}

__global__ void bias_sum_kernel(const float* __restrict__ a, const float* __restrict__ b,
                                float* __restrict__ dst, int n) {
  for (int i = blockIdx.x * blockDim.x + threadIdx.x; i < n; i += gridDim.x * blockDim.x)
    dst[i] = a[i] + b[i];
}

// xs[t*B + b][e] = emb[text[b][t]][e]  (f16)
__global__ void embed_kernel(const int* __restrict__ text, const float* __restrict__ emb,
                             _Float16* __restrict__ xf) {
  const int total = T_ * B_ * E_;
  for (int idx = blockIdx.x * blockDim.x + threadIdx.x; idx < total;
       idx += gridDim.x * blockDim.x) {
    int e = idx & (E_ - 1);
    int m = idx >> 8;           // m = t*32 + b
    int t = m >> 5, b = m & 31;
    int tok = text[b * T_ + t];
    xf[idx] = (_Float16)emb[(size_t)tok * E_ + e];
  }
}

// ---------------------------------------------------------------------------
// Gx[dir][t*32+b][n] = xs @ Wih_dir.T + (bih+bhh)   [8192 x 1024], K = 256
// ---------------------------------------------------------------------------
__global__ __launch_bounds__(256) void gx_gemm_kernel(
    const _Float16* __restrict__ xf,
    const _Float16* __restrict__ Wf, const _Float16* __restrict__ Wb,
    const float* __restrict__ biasf, const float* __restrict__ biasb,
    float* __restrict__ Gxf, float* __restrict__ Gxb) {
  const _Float16* W = blockIdx.z ? Wb : Wf;
  const float* bias = blockIdx.z ? biasb : biasf;
  float* out        = blockIdx.z ? Gxb : Gxf;
  int w = threadIdx.x >> 5, ln = threadIdx.x & 31;
  int mt = w & 1, nt4 = w >> 1;
  int ntile = blockIdx.y * 4 + nt4;
  int rowA = blockIdx.x * 32 + mt * 16 + (ln & 15);
  int kb0 = (ln >> 4) << 3;
  v8f acc = {};
#pragma unroll
  for (int kt = 0; kt < 8; ++kt) {
    v16h a = afrag_f16(xf + (size_t)rowA * E_, kt * 32 + kb0);
    v16h b = load_bfrag(W, 64, kt, ntile, ln);
    acc = wmma16(a, b, acc);
  }
  int n = ntile * 16 + (ln & 15);
  float bn = bias[n];
  int rbase = blockIdx.x * 32 + mt * 16 + ((ln & 16) ? 8 : 0);
#pragma unroll
  for (int r = 0; r < 8; ++r)
    out[(size_t)(rbase + r) * 1024 + n] = acc[r] + bn;
}

// ---------------------------------------------------------------------------
// Encoder LSTM: one 32-wave workgroup per direction. h in LDS f32, c in VGPRs.
// Wave w: mt=w&1 (row half), cg=w>>1 (hidden col group); owns i/f/g/o tiles at
// the SAME hidden column -> gate fusion fully in registers.
// ---------------------------------------------------------------------------
__global__ __launch_bounds__(1024) void enc_lstm_kernel(
    const _Float16* __restrict__ Whh_f, const _Float16* __restrict__ Whh_b,
    const float* __restrict__ Gxf, const float* __restrict__ Gxb,
    float* __restrict__ enc_out) {
  __shared__ float hbuf[B_][HE_];
  int dir = blockIdx.x;
  const _Float16* Whh = dir ? Whh_b : Whh_f;
  const float* Gx     = dir ? Gxb : Gxf;
  int colOfs = dir ? HE_ : 0;
  int w = threadIdx.x >> 5, ln = threadIdx.x & 31;
  int mt = w & 1, cg = w >> 1;
  int col = cg * 16 + (ln & 15);
  int rowA = mt * 16 + (ln & 15);
  int rbase = mt * 16 + ((ln & 16) ? 8 : 0);
  int kb0 = (ln >> 4) << 3;
  float cr[8];
#pragma unroll
  for (int r = 0; r < 8; ++r) cr[r] = 0.f;
  for (int i = threadIdx.x; i < B_ * HE_; i += 1024) (&hbuf[0][0])[i] = 0.f;
  __syncthreads();

  for (int tt = 0; tt < T_; ++tt) {
    int t = dir ? (T_ - 1 - tt) : tt;
    v8f acc[4] = {};
    for (int kt = 0; kt < 8; ++kt) {
      v16h a = afrag_f32(&hbuf[rowA][0], kt * 32 + kb0);
#pragma unroll
      for (int g = 0; g < 4; ++g)
        acc[g] = wmma16(a, load_bfrag(Whh, 64, kt, g * 16 + cg, ln), acc[g]);
    }
    __syncthreads();  // all reads of hbuf complete
    float hn8[8];
#pragma unroll
    for (int r = 0; r < 8; ++r) {
      int row = rbase + r;
      const float* gp = Gx + ((size_t)(t * B_ + row)) * 1024;
      float pi = acc[0][r] + gp[0 * HE_ + col];
      float pf = acc[1][r] + gp[1 * HE_ + col];
      float pg = acc[2][r] + gp[2 * HE_ + col];
      float po = acc[3][r] + gp[3 * HE_ + col];
      float cn = sigm(pf) * cr[r] + sigm(pi) * tanh_fast(pg);
      cr[r] = cn;
      hn8[r] = sigm(po) * tanh_fast(cn);
    }
#pragma unroll
    for (int r = 0; r < 8; ++r) {
      int row = rbase + r;
      hbuf[row][col] = hn8[r];
      enc_out[((size_t)row * T_ + t) * H_ + colOfs + col] = hn8[r];
    }
    __syncthreads();
  }
}

// --- tiny speaker MLP ---
__global__ void spk1_kernel(const float* __restrict__ se, const float* __restrict__ W1,
                            const float* __restrict__ b1, float* __restrict__ o) {
  for (int i = blockIdx.x * blockDim.x + threadIdx.x; i < B_ * H_;
       i += gridDim.x * blockDim.x) {
    int b = i >> 9, n = i & (H_ - 1);
    float s = b1[n];
    for (int k = 0; k < S_; ++k) s += se[b * S_ + k] * W1[(size_t)n * S_ + k];
    o[i] = fmaxf(s, 0.f);
  }
}
__global__ void spk2_kernel(const float* __restrict__ x, const float* __restrict__ W2,
                            const float* __restrict__ b2, float* __restrict__ o) {
  for (int i = blockIdx.x * blockDim.x + threadIdx.x; i < B_ * H_;
       i += gridDim.x * blockDim.x) {
    int b = i >> 9, n = i & (H_ - 1);
    float s = b2[n];
    for (int k = 0; k < H_; ++k) s += x[b * H_ + k] * W2[(size_t)n * H_ + k];
    o[i] = s;
  }
}
__global__ void spk_add_kernel(float* __restrict__ enc_out, const float* __restrict__ spk) {
  const int total = B_ * T_ * H_;
  for (int i = blockIdx.x * blockDim.x + threadIdx.x; i < total;
       i += gridDim.x * blockDim.x) {
    int b = i / (T_ * H_);
    int h = i & (H_ - 1);
    enc_out[i] += spk[b * H_ + h];
  }
}

// ---------------------------------------------------------------------------
// Autoregressive decoder: one 32-wave workgroup, max_len steps.
// LDS: melq[32][96] | qbuf[32][512] | h0b[32][512] | h1b[32][512] | melh[32][512]
// Layer-0 K stack = [mel|pad(96)] + q(512) + h0(512) -> 35 k-tiles vs packed
// [Wih0(mel) ; Wih0(q) ; Whh0].  q tile staged by the Tensor Data Mover,
// prefetched one step ahead (overlaps layer-1 + mel GEMMs); wave 0 drains it
// with s_wait_tensorcnt before the end-of-step barrier.
// c0/c1 live in VGPRs for all 500 steps.
// ---------------------------------------------------------------------------
__global__ __launch_bounds__(1024) void dec_kernel(
    const _Float16* __restrict__ Wcat0, const float* __restrict__ bdec0,
    const _Float16* __restrict__ Wcat1, const float* __restrict__ bdec1,
    const _Float16* __restrict__ Wmel1, const float* __restrict__ mel_b1,
    const _Float16* __restrict__ Wmel2, const float* __restrict__ mel_b2,
    const float* __restrict__ stop_W, const float* __restrict__ stop_b,
    const float* __restrict__ enc_out, const int* __restrict__ max_len_p,
    float* __restrict__ out) {
  extern __shared__ float lds[];
  float* melq = lds;                       // [32][96]  (mel 0..79 | pad)
  float* qb   = melq + 32 * 96;            // [32][512]
  float* h0b  = qb   + 32 * 512;           // [32][512]
  float* h1b  = h0b  + 32 * 512;           // [32][512]
  float* melh = h1b  + 32 * 512;           // [32][512]
  const int max_len = max_len_p[0];
  float* mel_out  = out;                               // [32][max_len][80]
  float* stop_out = out + (size_t)B_ * max_len * M_;   // [32][max_len]

  int tid = threadIdx.x;
  int w = tid >> 5, ln = tid & 31;
  int mt = w & 1;
  int rowA  = mt * 16 + (ln & 15);
  int rbase = mt * 16 + ((ln & 16) ? 8 : 0);
  int kb0 = (ln >> 4) << 3;
  int lcol = ln & 15;

  float c0r[2][8], c1r[2][8];
#pragma unroll
  for (int p = 0; p < 2; ++p)
#pragma unroll
    for (int r = 0; r < 8; ++r) { c0r[p][r] = 0.f; c1r[p][r] = 0.f; }

  for (int i = tid; i < 32 * 96; i += 1024)  melq[i] = 0.f;
  for (int i = tid; i < 32 * 512; i += 1024) { h0b[i] = 0.f; h1b[i] = 0.f; }
  unsigned qb_lds = (unsigned)(size_t)(void*)qb;   // LDS byte address (low 32 bits)

  // TDM-stage q for step 0: 32 rows x 512 f32, row stride T*H elements.
  if (w == 0) {
    tdm_load_2d(qb_lds, enc_out /* + tq0*H, tq0==0 */, H_, B_, T_ * H_);
    __builtin_amdgcn_s_wait_tensorcnt(0);
  }
  __syncthreads();

  for (int step = 0; step < max_len; ++step) {
    // ---- layer 0: gates[32,2048] = [mel|q|h0] @ [Wih0;Whh0]  (K=35 k-tiles)
    float h0v[2][8];
#pragma unroll
    for (int p = 0; p < 2; ++p) {
      int cg = (w >> 1) * 2 + p;     // 0..31
      v8f acc[4] = {};
      for (int kt = 0; kt < 35; ++kt) {
        if (kt + 1 < 35)
          __builtin_prefetch(Wcat0 + (((size_t)(kt + 1) * 128 + cg) << 9), 0, 1);
        const float* abase;
        int kk;
        if (kt < 3)       { abase = melq + (size_t)rowA * 96;  kk = kt * 32 + kb0; }
        else if (kt < 19) { abase = qb   + (size_t)rowA * 512; kk = (kt - 3) * 32 + kb0; }
        else              { abase = h0b  + (size_t)rowA * 512; kk = (kt - 19) * 32 + kb0; }
        v16h a = afrag_f32(abase, kk);
#pragma unroll
        for (int g = 0; g < 4; ++g)
          acc[g] = wmma16(a, load_bfrag(Wcat0, 128, kt, g * 32 + cg, ln), acc[g]);
      }
      int col = cg * 16 + lcol;      // 0..511
#pragma unroll
      for (int r = 0; r < 8; ++r) {
        float pi = acc[0][r] + bdec0[0 * H_ + col];
        float pf = acc[1][r] + bdec0[1 * H_ + col];
        float pg = acc[2][r] + bdec0[2 * H_ + col];
        float po = acc[3][r] + bdec0[3 * H_ + col];
        float cn = sigm(pf) * c0r[p][r] + sigm(pi) * tanh_fast(pg);
        c0r[p][r] = cn;
        h0v[p][r] = sigm(po) * tanh_fast(cn);
      }
    }
    __syncthreads();  // all reads of melq/qb/h0b done

    // qb free: TDM-prefetch next step's query (overlaps layer 1 + mel stages)
    if (w == 0 && step + 1 < max_len) {
      int tqn = (step + 1) & (T_ - 1);
      tdm_load_2d(qb_lds, enc_out + (size_t)tqn * H_, H_, B_, T_ * H_);
    }
#pragma unroll
    for (int p = 0; p < 2; ++p) {
      int col = ((w >> 1) * 2 + p) * 16 + lcol;
#pragma unroll
      for (int r = 0; r < 8; ++r)
        h0b[(size_t)(rbase + r) * 512 + col] = h0v[p][r];
    }
    __syncthreads();

    // ---- layer 1: gates[32,2048] = [h0|h1] @ [Wih1;Whh1]  (K=1024)
    float h1v[2][8];
#pragma unroll
    for (int p = 0; p < 2; ++p) {
      int cg = (w >> 1) * 2 + p;
      v8f acc[4] = {};
      for (int kt = 0; kt < 32; ++kt) {
        const float* abase = (kt < 16) ? (h0b + (size_t)rowA * 512)
                                       : (h1b + (size_t)rowA * 512);
        int kk = ((kt < 16) ? kt : (kt - 16)) * 32 + kb0;
        v16h a = afrag_f32(abase, kk);
#pragma unroll
        for (int g = 0; g < 4; ++g)
          acc[g] = wmma16(a, load_bfrag(Wcat1, 128, kt, g * 32 + cg, ln), acc[g]);
      }
      int col = cg * 16 + lcol;
#pragma unroll
      for (int r = 0; r < 8; ++r) {
        float pi = acc[0][r] + bdec1[0 * H_ + col];
        float pf = acc[1][r] + bdec1[1 * H_ + col];
        float pg = acc[2][r] + bdec1[2 * H_ + col];
        float po = acc[3][r] + bdec1[3 * H_ + col];
        float cn = sigm(pf) * c1r[p][r] + sigm(pi) * tanh_fast(pg);
        c1r[p][r] = cn;
        h1v[p][r] = sigm(po) * tanh_fast(cn);
      }
    }
    __syncthreads();  // reads of old h1 done
#pragma unroll
    for (int p = 0; p < 2; ++p) {
      int col = ((w >> 1) * 2 + p) * 16 + lcol;
#pragma unroll
      for (int r = 0; r < 8; ++r)
        h1b[(size_t)(rbase + r) * 512 + col] = h1v[p][r];
    }
    __syncthreads();

    // ---- melh = relu(h1 @ mel_W1.T + b1)   [32,512], K=512
    {
      int ntA = w >> 1;              // 0..15
      v8f a0 = {}, a1 = {};
      for (int kt = 0; kt < 16; ++kt) {
        v16h a = afrag_f32(h1b + (size_t)rowA * 512, kt * 32 + kb0);
        a0 = wmma16(a, load_bfrag(Wmel1, 32, kt, ntA, ln), a0);
        a1 = wmma16(a, load_bfrag(Wmel1, 32, kt, ntA + 16, ln), a1);
      }
      int n0 = ntA * 16 + lcol, n1 = n0 + 256;
#pragma unroll
      for (int r = 0; r < 8; ++r) {
        melh[(size_t)(rbase + r) * 512 + n0] = fmaxf(a0[r] + mel_b1[n0], 0.f);
        melh[(size_t)(rbase + r) * 512 + n1] = fmaxf(a1[r] + mel_b1[n1], 0.f);
      }
    }
    __syncthreads();

    // ---- mel = melh @ mel_W2.T + b2  [32,80] (waves 0..9); stop = h1 @ stop_W
    if (w < 10) {
      int mt2 = w & 1, nt2 = w >> 1;           // nt2: 0..4
      int rowA2 = mt2 * 16 + (ln & 15);
      int rb2 = mt2 * 16 + ((ln & 16) ? 8 : 0);
      v8f acc = {};
      for (int kt = 0; kt < 16; ++kt)
        acc = wmma16(afrag_f32(melh + (size_t)rowA2 * 512, kt * 32 + kb0),
                     load_bfrag(Wmel2, 5, kt, nt2, ln), acc);
      int n = nt2 * 16 + lcol;                 // < 80
      float bn = mel_b2[n];
#pragma unroll
      for (int r = 0; r < 8; ++r) {
        int row = rb2 + r;
        float v = acc[r] + bn;
        mel_out[((size_t)row * max_len + step) * M_ + n] = v;
        melq[(size_t)row * 96 + n] = v;        // dec_in for next step
      }
    }
    if (tid < 32) {                            // wave 0: stop logits
      float s = stop_b[0];
      for (int j = 0; j < H_; ++j) s += h1b[(size_t)tid * 512 + j] * stop_W[j];
      stop_out[(size_t)tid * max_len + step] = s;
    }
    if (w == 0) __builtin_amdgcn_s_wait_tensorcnt(0);  // next q landed in LDS
    __syncthreads();
  }
}

// ---------------------------------------------------------------------------
extern "C" void kernel_launch(void* const* d_in, const int* in_sizes, int n_in,
                              void* d_out, int out_size, void* d_ws, size_t ws_size,
                              hipStream_t stream) {
  (void)in_sizes; (void)n_in; (void)out_size; (void)ws_size;
  const int*   text    = (const int*)  d_in[0];
  const float* spk_emb = (const float*)d_in[1];
  const float* emb     = (const float*)d_in[2];
  const float* eWih_f  = (const float*)d_in[3];
  const float* eWhh_f  = (const float*)d_in[4];
  const float* ebih_f  = (const float*)d_in[5];
  const float* ebhh_f  = (const float*)d_in[6];
  const float* eWih_b  = (const float*)d_in[7];
  const float* eWhh_b  = (const float*)d_in[8];
  const float* ebih_b  = (const float*)d_in[9];
  const float* ebhh_b  = (const float*)d_in[10];
  const float* spk_W1  = (const float*)d_in[11];
  const float* spk_b1  = (const float*)d_in[12];
  const float* spk_W2  = (const float*)d_in[13];
  const float* spk_b2  = (const float*)d_in[14];
  const float* dWih0   = (const float*)d_in[15];
  const float* dWhh0   = (const float*)d_in[16];
  const float* dbih0   = (const float*)d_in[17];
  const float* dbhh0   = (const float*)d_in[18];
  const float* dWih1   = (const float*)d_in[19];
  const float* dWhh1   = (const float*)d_in[20];
  const float* dbih1   = (const float*)d_in[21];
  const float* dbhh1   = (const float*)d_in[22];
  const float* mel_W1  = (const float*)d_in[23];
  const float* mel_b1  = (const float*)d_in[24];
  const float* mel_W2  = (const float*)d_in[25];
  const float* mel_b2  = (const float*)d_in[26];
  const float* stop_W  = (const float*)d_in[27];
  const float* stop_b  = (const float*)d_in[28];
  const int*   max_len = (const int*)  d_in[29];

  // workspace layout
  char* ws = (char*)d_ws;
  size_t o = 0;
  auto alloc = [&](size_t bytes) { size_t r = o; o = (o + bytes + 1023) & ~(size_t)1023; return r; };
  size_t oWencIHf = alloc((size_t)8  * 64  * 512 * 2);
  size_t oWencHHf = alloc((size_t)8  * 64  * 512 * 2);
  size_t oWencIHb = alloc((size_t)8  * 64  * 512 * 2);
  size_t oWencHHb = alloc((size_t)8  * 64  * 512 * 2);
  size_t oWcat0   = alloc((size_t)35 * 128 * 512 * 2);
  size_t oWcat1   = alloc((size_t)32 * 128 * 512 * 2);
  size_t oWmel1   = alloc((size_t)16 * 32  * 512 * 2);
  size_t oWmel2   = alloc((size_t)16 * 5   * 512 * 2);
  size_t oBencF   = alloc(1024 * 4);
  size_t oBencB   = alloc(1024 * 4);
  size_t oBdec0   = alloc(2048 * 4);
  size_t oBdec1   = alloc(2048 * 4);
  size_t zero_end = o;
  size_t oXf      = alloc((size_t)8192 * 256 * 2);
  size_t oGxf     = alloc((size_t)8192 * 1024 * 4);
  size_t oGxb     = alloc((size_t)8192 * 1024 * 4);
  size_t oEnc     = alloc((size_t)B_ * T_ * H_ * 4);
  size_t oSpk1    = alloc((size_t)B_ * H_ * 4);
  size_t oSpk2    = alloc((size_t)B_ * H_ * 4);

  _Float16* WencIHf = (_Float16*)(ws + oWencIHf);
  _Float16* WencHHf = (_Float16*)(ws + oWencHHf);
  _Float16* WencIHb = (_Float16*)(ws + oWencIHb);
  _Float16* WencHHb = (_Float16*)(ws + oWencHHb);
  _Float16* Wcat0   = (_Float16*)(ws + oWcat0);
  _Float16* Wcat1   = (_Float16*)(ws + oWcat1);
  _Float16* Wmel1   = (_Float16*)(ws + oWmel1);
  _Float16* Wmel2   = (_Float16*)(ws + oWmel2);
  float* bencF = (float*)(ws + oBencF);
  float* bencB = (float*)(ws + oBencB);
  float* bdec0 = (float*)(ws + oBdec0);
  float* bdec1 = (float*)(ws + oBdec1);
  _Float16* xf = (_Float16*)(ws + oXf);
  float* Gxf  = (float*)(ws + oGxf);
  float* Gxb  = (float*)(ws + oGxb);
  float* enco = (float*)(ws + oEnc);
  float* spk1 = (float*)(ws + oSpk1);
  float* spk2 = (float*)(ws + oSpk2);

  (void)hipMemsetAsync(d_ws, 0, zero_end, stream);

  // --- pack weights into WMMA B-fragment tiles (f16) ---
  dim3 pk(2048), pt(256);
  pack_b_kernel<<<pk, pt, 0, stream>>>(eWih_f, 1024, 256, 0, 256, WencIHf, 64, 0);
  pack_b_kernel<<<pk, pt, 0, stream>>>(eWhh_f, 1024, 256, 0, 256, WencHHf, 64, 0);
  pack_b_kernel<<<pk, pt, 0, stream>>>(eWih_b, 1024, 256, 0, 256, WencIHb, 64, 0);
  pack_b_kernel<<<pk, pt, 0, stream>>>(eWhh_b, 1024, 256, 0, 256, WencHHb, 64, 0);
  // layer-0 stacked K: [Wih0 mel cols 0..79 | Wih0 q cols 80..591 | Whh0]
  pack_b_kernel<<<pk, pt, 0, stream>>>(dWih0, 2048, 592, 0,  80,  Wcat0, 128, 0);
  pack_b_kernel<<<pk, pt, 0, stream>>>(dWih0, 2048, 592, 80, 512, Wcat0, 128, 3);
  pack_b_kernel<<<pk, pt, 0, stream>>>(dWhh0, 2048, 512, 0,  512, Wcat0, 128, 19);
  pack_b_kernel<<<pk, pt, 0, stream>>>(dWih1, 2048, 512, 0,  512, Wcat1, 128, 0);
  pack_b_kernel<<<pk, pt, 0, stream>>>(dWhh1, 2048, 512, 0,  512, Wcat1, 128, 16);
  pack_b_kernel<<<pk, pt, 0, stream>>>(mel_W1, 512, 512, 0,  512, Wmel1, 32, 0);
  pack_b_kernel<<<pk, pt, 0, stream>>>(mel_W2,  80, 512, 0,  512, Wmel2, 5, 0);

  bias_sum_kernel<<<8, 256, 0, stream>>>(ebih_f, ebhh_f, bencF, 1024);
  bias_sum_kernel<<<8, 256, 0, stream>>>(ebih_b, ebhh_b, bencB, 1024);
  bias_sum_kernel<<<8, 256, 0, stream>>>(dbih0,  dbhh0,  bdec0, 2048);
  bias_sum_kernel<<<8, 256, 0, stream>>>(dbih1,  dbhh1,  bdec1, 2048);

  // --- embedding (f16) + input-projection GEMMs for both directions ---
  embed_kernel<<<2048, 256, 0, stream>>>(text, emb, xf);
  gx_gemm_kernel<<<dim3(256, 16, 2), 256, 0, stream>>>(xf, WencIHf, WencIHb,
                                                       bencF, bencB, Gxf, Gxb);

  // --- bidirectional encoder recurrence (2 workgroups, 32 waves each) ---
  enc_lstm_kernel<<<2, 1024, 0, stream>>>(WencHHf, WencHHb, Gxf, Gxb, enco);

  // --- speaker MLP + broadcast add ---
  spk1_kernel<<<64, 256, 0, stream>>>(spk_emb, spk_W1, spk_b1, spk1);
  spk2_kernel<<<64, 256, 0, stream>>>(spk1, spk_W2, spk_b2, spk2);
  spk_add_kernel<<<4096, 256, 0, stream>>>(enco, spk2);

  // --- autoregressive decoder (1 workgroup, 32 waves, 274KB LDS, TDM q-feed) ---
  size_t dec_lds = (size_t)(32 * 96 + 4 * 32 * 512) * sizeof(float);
  dec_kernel<<<1, 1024, dec_lds, stream>>>(Wcat0, bdec0, Wcat1, bdec1,
                                           Wmel1, mel_b1, Wmel2, mel_b2,
                                           stop_W, stop_b, enco, max_len,
                                           (float*)d_out);
}
